// GraphAttentionLayer_85650237817501
// MI455X (gfx1250) — matrix operand order repaired and checked by reading, not compiled, not run on hardware
//
#include <hip/hip_runtime.h>

#define BGRAPH 1024
#define MAXLEN 64
#define EDIM   256
#define NHEAD  8
#define HDIM   32
#define NEGV   -1e9f

typedef __attribute__((ext_vector_type(16))) __bf16       v16bf;
typedef __attribute__((ext_vector_type(2)))  __bf16       bf16x2;
typedef __attribute__((ext_vector_type(8)))  float        v8f;
typedef __attribute__((ext_vector_type(2)))  float        f32x2;
typedef __attribute__((ext_vector_type(4)))  float        f32x4;
typedef __attribute__((ext_vector_type(2)))  unsigned int u32x2;
typedef __attribute__((ext_vector_type(4)))  unsigned int u32x4;

union Frag  { v16bf v; u32x4 q[2]; };

// Hardware bf16 converts (v_cvt_pk_bf16_f32 path)
__device__ __forceinline__ unsigned short f2bf(float x) {
  __bf16 b = (__bf16)x;
  return __builtin_bit_cast(unsigned short, b);
}
__device__ __forceinline__ unsigned int pk_bf16(float a, float b) {
  f32x2 f = {a, b};
  bf16x2 h = __builtin_convertvector(f, bf16x2);
  return __builtin_bit_cast(unsigned int, h);
}

// ---- LDS layout (bytes). X region is reused for CTX after GEMM1. ----
#define XS_OFF   0            // X  bf16 [64][272]   34,816 B  (CTX [64][264] later)
#define QK_OFF   34816        // QK bf16 [64][520]   66,560 B  (Q cols 0..255, K cols 256..511)
#define VT_OFF   101376       // Vt bf16 [256][72]   36,864 B  (V transposed: [d][key])
#define AT_OFF   138240       // At bf16 [8][64][72] 73,728 B  (attention per head)
#define SMEM_BYTES 211968

#define XS_STRIDE 272
#define QK_STRIDE 520
#define VT_STRIDE 72
#define AT_STRIDE 72
#define CT_STRIDE 264

__global__ void convert_weights_kernel(const float* __restrict__ wq,
                                       const float* __restrict__ wo,
                                       unsigned int* __restrict__ dq,
                                       unsigned int* __restrict__ dwo) {
  const int nq2 = 3 * EDIM * EDIM / 2;   // 98304 pairs
  const int no2 = EDIM * EDIM / 2;       // 32768 pairs
  for (int i = blockIdx.x * blockDim.x + threadIdx.x; i < nq2 + no2;
       i += gridDim.x * blockDim.x) {
    if (i < nq2) {
      f32x2 v = *(const f32x2*)(wq + 2 * (size_t)i);
      dq[i] = pk_bf16(v.x, v.y);
    } else {
      int j = i - nq2;
      f32x2 v = *(const f32x2*)(wo + 2 * (size_t)j);
      dwo[j] = pk_bf16(v.x, v.y);
    }
  }
}

// batch_indices is sorted: boundary detection gives starts[] without atomics.
__global__ void find_starts_kernel(const int* __restrict__ bidx,
                                   int* __restrict__ starts, int N) {
  int i = blockIdx.x * blockDim.x + threadIdx.x;
  if (i >= N) return;
  int b = bidx[i];
  if (i == 0 || bidx[i - 1] != b) starts[b] = i;
  if (i == N - 1) starts[BGRAPH] = N;
}

__global__ __launch_bounds__(256, 1)
void gat_attn_kernel(const float* __restrict__ nf,
                     const int* __restrict__ starts,
                     const float* __restrict__ inb,
                     const float* __restrict__ outb,
                     const unsigned short* __restrict__ wqkv,   // bf16 [768][256]
                     const unsigned short* __restrict__ wo,     // bf16 [256][256]
                     float* __restrict__ out,                   // [64][1024][256]
                     float* __restrict__ wts)                   // [1024][64][64]
{
  extern __shared__ unsigned char smem[];
  unsigned short* Xs = (unsigned short*)(smem + XS_OFF);
  unsigned short* CT = Xs;                                   // reuse after GEMM1
  unsigned short* QK = (unsigned short*)(smem + QK_OFF);
  unsigned short* Vt = (unsigned short*)(smem + VT_OFF);
  unsigned short* At = (unsigned short*)(smem + AT_OFF);

  const int g     = blockIdx.x;
  const int tid   = threadIdx.x;
  const int wave  = tid >> 5;
  const int lane  = tid & 31;
  const int laneM = lane & 15;   // column / row-within-tile index
  const int laneG = lane >> 4;   // half-wave group

  const int start = starts[g];
  const int count = starts[g + 1] - start;

  // ---------------- Phase 0: load X, f32 -> bf16, zero-pad rows >= count ----
  for (int idx = tid; idx < MAXLEN * (EDIM / 4); idx += 256) {
    int row = idx >> 6;
    int c4  = idx & 63;
    f32x4 v = {0.f, 0.f, 0.f, 0.f};
    if (row < count) v = *(const f32x4*)(nf + (size_t)(start + row) * EDIM + c4 * 4);
    u32x2 p = { pk_bf16(v.x, v.y), pk_bf16(v.z, v.w) };
    *(u32x2*)(Xs + row * XS_STRIDE + c4 * 4) = p;
  }
  __syncthreads();

  // ---------------- Phase 1: QKV = X @ Wqkv^T + b ----
  // A-fragment (X) loader + 8-step K accumulation, shared by both tile groups.
  auto gemm1_acc = [&](int mt, const Frag* bfr, float bias) -> v8f {
    v8f acc;
#pragma unroll
    for (int r = 0; r < 8; ++r) acc[r] = bias;
    const unsigned short* arow = Xs + (mt * 16 + laneM) * XS_STRIDE;
#pragma unroll
    for (int ks = 0; ks < 8; ++ks) {
      Frag a;                                               // A layout: K {0-15}/{16-31}
      a.q[0] = *(const u32x4*)(arow + ks * 32 + laneG * 16);
      a.q[1] = *(const u32x4*)(arow + ks * 32 + laneG * 16 + 8);
      acc = __builtin_amdgcn_wmma_f32_16x16x32_bf16(false, a.v, false, bfr[ks].v,
                                                    (short)0, acc, false, false);
    }
    return acc;
  };

  // --- Q/K tiles: nt = wave*4 + i  (0..31), row-major store path only ---
#pragma unroll
  for (int i = 0; i < 4; ++i) {
    const int nt   = wave * 4 + i;
    const int ncol = nt * 16 + laneM;                       // 0..511
    Frag bfr[8];
    const u32x4* wrow = (const u32x4*)(wqkv + (size_t)ncol * EDIM);
#pragma unroll
    for (int ks = 0; ks < 8; ++ks) {                        // 32B contiguous per lane
      bfr[ks].q[0] = wrow[ks * 4 + laneG * 2];
      bfr[ks].q[1] = wrow[ks * 4 + laneG * 2 + 1];
    }
    const float bias = inb[ncol];
#pragma unroll
    for (int mt = 0; mt < 4; ++mt) {
      v8f acc = gemm1_acc(mt, bfr, bias);
#pragma unroll
      for (int r = 0; r < 8; ++r)
        QK[(mt * 16 + r + 8 * laneG) * QK_STRIDE + ncol] = f2bf(acc[r]);
    }
  }

  // --- V tiles: nt = 32 + wave*2 + i  (32..47), transposed packed store ---
#pragma unroll
  for (int i = 0; i < 2; ++i) {
    const int nt   = 32 + wave * 2 + i;
    const int ncol = nt * 16 + laneM;                       // 512..767
    Frag bfr[8];
    const u32x4* wrow = (const u32x4*)(wqkv + (size_t)ncol * EDIM);
#pragma unroll
    for (int ks = 0; ks < 8; ++ks) {
      bfr[ks].q[0] = wrow[ks * 4 + laneG * 2];
      bfr[ks].q[1] = wrow[ks * 4 + laneG * 2 + 1];
    }
    const float bias = inb[ncol];
#pragma unroll
    for (int mt = 0; mt < 4; ++mt) {
      v8f acc = gemm1_acc(mt, bfr, bias);
      u32x4 p = { pk_bf16(acc[0], acc[1]), pk_bf16(acc[2], acc[3]),
                  pk_bf16(acc[4], acc[5]), pk_bf16(acc[6], acc[7]) };
      *(u32x4*)(Vt + ((nt - 32) * 16 + laneM) * VT_STRIDE + mt * 16 + 8 * laneG) = p;
    }
  }
  __syncthreads();

  // ---------------- Phase 2: per-head scores + register softmax ----
  const int h = wave;
  const float scale = 0.17677669529663687f;                   // 1/sqrt(32)
#pragma unroll
  for (int mt = 0; mt < 4; ++mt) {
    Frag qa;
    const unsigned short* qrow = QK + (mt * 16 + laneM) * QK_STRIDE + h * HDIM;
    qa.q[0] = *(const u32x4*)(qrow + laneG * 16);
    qa.q[1] = *(const u32x4*)(qrow + laneG * 16 + 8);
    v8f s[4];
#pragma unroll
    for (int nt = 0; nt < 4; ++nt) {
      Frag kb;                                                // key row, contiguous d
      const unsigned short* krow = QK + (nt * 16 + laneM) * QK_STRIDE + EDIM + h * HDIM + laneG * 16;
      kb.q[0] = *(const u32x4*)(krow);
      kb.q[1] = *(const u32x4*)(krow + 8);
      v8f z = {};
      s[nt] = __builtin_amdgcn_wmma_f32_16x16x32_bf16(false, qa.v, false, kb.v,
                                                      (short)0, z, false, false);
    }
#pragma unroll
    for (int nt = 0; nt < 4; ++nt) {                          // scale then mask pad keys
      const bool pad = (nt * 16 + laneM) >= count;
#pragma unroll
      for (int r = 0; r < 8; ++r) s[nt][r] = pad ? NEGV : s[nt][r] * scale;
    }
#pragma unroll
    for (int r = 0; r < 8; ++r) {                             // softmax: row = vgpr r + 8*laneG
      float mx = fmaxf(fmaxf(s[0][r], s[1][r]), fmaxf(s[2][r], s[3][r]));
#pragma unroll
      for (int o = 8; o >= 1; o >>= 1) mx = fmaxf(mx, __shfl_xor(mx, o, 32));
      float sum = 0.f;
#pragma unroll
      for (int nt = 0; nt < 4; ++nt) { float e = __expf(s[nt][r] - mx); s[nt][r] = e; sum += e; }
#pragma unroll
      for (int o = 8; o >= 1; o >>= 1) sum += __shfl_xor(sum, o, 32);
      const float inv = 1.f / sum;
#pragma unroll
      for (int nt = 0; nt < 4; ++nt) s[nt][r] *= inv;
    }
#pragma unroll
    for (int nt = 0; nt < 4; ++nt)
#pragma unroll
      for (int r = 0; r < 8; ++r)
        At[h * (MAXLEN * AT_STRIDE) + (mt * 16 + r + 8 * laneG) * AT_STRIDE + nt * 16 + laneM]
            = f2bf(s[nt][r]);
  }
  __syncthreads();

  // ---------------- Phase 3: ctx = attn @ V  (per head, K=64 -> 2 k-steps) ----
#pragma unroll
  for (int mt = 0; mt < 4; ++mt) {
    const unsigned short* arow = At + h * (MAXLEN * AT_STRIDE) + (mt * 16 + laneM) * AT_STRIDE;
    Frag a0, a1;
    a0.q[0] = *(const u32x4*)(arow + laneG * 16);
    a0.q[1] = *(const u32x4*)(arow + laneG * 16 + 8);
    a1.q[0] = *(const u32x4*)(arow + 32 + laneG * 16);
    a1.q[1] = *(const u32x4*)(arow + 32 + laneG * 16 + 8);
#pragma unroll
    for (int ntd = 0; ntd < 2; ++ntd) {
      const unsigned short* vrow = Vt + (h * HDIM + ntd * 16 + laneM) * VT_STRIDE;
      Frag b0, b1;                                            // Vt row = contiguous keys
      b0.q[0] = *(const u32x4*)(vrow + laneG * 16);
      b0.q[1] = *(const u32x4*)(vrow + laneG * 16 + 8);
      b1.q[0] = *(const u32x4*)(vrow + 32 + laneG * 16);
      b1.q[1] = *(const u32x4*)(vrow + 32 + laneG * 16 + 8);
      v8f acc = {};
      acc = __builtin_amdgcn_wmma_f32_16x16x32_bf16(false, a0.v, false, b0.v, (short)0, acc, false, false);
      acc = __builtin_amdgcn_wmma_f32_16x16x32_bf16(false, a1.v, false, b1.v, (short)0, acc, false, false);
      const int col = h * HDIM + ntd * 16 + laneM;
#pragma unroll
      for (int r = 0; r < 8; ++r)
        CT[(mt * 16 + r + 8 * laneG) * CT_STRIDE + col] = f2bf(acc[r]);
    }
  }
  __syncthreads();

  // ---------------- Phase 4: out = ctx @ Wout^T + b, write [L,B,E] ----
#pragma unroll
  for (int i = 0; i < 2; ++i) {
    const int nt = wave + i * 8;
    const int e  = nt * 16 + laneM;
    Frag bw[8];
    const u32x4* wrow = (const u32x4*)(wo + (size_t)e * EDIM);
#pragma unroll
    for (int ks = 0; ks < 8; ++ks) {
      bw[ks].q[0] = wrow[ks * 4 + laneG * 2];
      bw[ks].q[1] = wrow[ks * 4 + laneG * 2 + 1];
    }
    const float bias = outb[e];
#pragma unroll
    for (int mt = 0; mt < 4; ++mt) {
      v8f acc;
#pragma unroll
      for (int r = 0; r < 8; ++r) acc[r] = bias;
      const unsigned short* arow = CT + (mt * 16 + laneM) * CT_STRIDE;
#pragma unroll
      for (int ks = 0; ks < 8; ++ks) {
        Frag a;
        a.q[0] = *(const u32x4*)(arow + ks * 32 + laneG * 16);
        a.q[1] = *(const u32x4*)(arow + ks * 32 + laneG * 16 + 8);
        acc = __builtin_amdgcn_wmma_f32_16x16x32_bf16(false, a.v, false, bw[ks].v,
                                                      (short)0, acc, false, false);
      }
#pragma unroll
      for (int r = 0; r < 8; ++r) {
        const int m = mt * 16 + r + 8 * laneG;
        out[(size_t)m * (BGRAPH * EDIM) + (size_t)g * EDIM + e] = acc[r];
      }
    }
  }

  // ---------------- Phase 5: weights = mean over heads of attn ----
  // Vectorized: each iteration handles a pair of adjacent key columns.
  for (int idx = tid; idx < MAXLEN * (MAXLEN / 2); idx += 256) {
    const int l = idx >> 5, m2 = idx & 31;                    // m = 2*m2, 2*m2+1
    float s0 = 0.f, s1 = 0.f;
#pragma unroll
    for (int hh = 0; hh < NHEAD; ++hh) {
      unsigned int p = *(const unsigned int*)
          (At + hh * (MAXLEN * AT_STRIDE) + l * AT_STRIDE + m2 * 2);
      s0 += __uint_as_float(p << 16);
      s1 += __uint_as_float(p & 0xFFFF0000u);
    }
    f32x2 o = { s0 * 0.125f, s1 * 0.125f };
    *(f32x2*)(wts + (size_t)g * (MAXLEN * MAXLEN) + l * MAXLEN + m2 * 2) = o;
  }
}

extern "C" void kernel_launch(void* const* d_in, const int* in_sizes, int n_in,
                              void* d_out, int out_size, void* d_ws, size_t ws_size,
                              hipStream_t stream) {
  const float* nf   = (const float*)d_in[0];
  const int*   bidx = (const int*)d_in[1];
  const float* inw  = (const float*)d_in[2];
  const float* inb  = (const float*)d_in[3];
  const float* outw = (const float*)d_in[4];
  const float* outb = (const float*)d_in[5];
  float* out = (float*)d_out;
  float* wts = out + (size_t)MAXLEN * BGRAPH * EDIM;
  const int N = in_sizes[1];

  unsigned short* wq = (unsigned short*)d_ws;
  unsigned short* wo = wq + 3 * EDIM * EDIM;
  int* starts = (int*)((char*)d_ws +
                       (size_t)(3 * EDIM * EDIM + EDIM * EDIM) * sizeof(unsigned short));

  convert_weights_kernel<<<256, 256, 0, stream>>>(inw, outw,
                                                  (unsigned int*)wq, (unsigned int*)wo);
  find_starts_kernel<<<(N + 255) / 256, 256, 0, stream>>>(bidx, starts, N);

  hipFuncSetAttribute((const void*)gat_attn_kernel,
                      hipFuncAttributeMaxDynamicSharedMemorySize, SMEM_BYTES);
  gat_attn_kernel<<<BGRAPH, 256, SMEM_BYTES, stream>>>(nf, starts, inb, outb,
                                                       wq, wo, out, wts);
}
